// GDNSequenceModel_48206712930748
// MI455X (gfx1250) — compile-verified, matrix operand order checked by reading, not run on hardware
//
#include <hip/hip_runtime.h>
#include <cstdint>

// ---------------------------------------------------------------------------
// Gated DeltaNet forward for MI455X (gfx1250, wave32, WMMA).
// B=4, T=4096, D=1024, H=4, DK=DV=256, chunk L=64, C=64 chunks.
// ---------------------------------------------------------------------------

typedef __attribute__((ext_vector_type(16))) __bf16 v16bf;
typedef __attribute__((ext_vector_type(8)))  __bf16 v8bf;
typedef __attribute__((ext_vector_type(8)))  float  v8f;

#define GB 4
#define GT 4096
#define GD 1024
#define GH 4
#define GDK 256
#define GDV 256
#define CHL 64          // chunk length
#define NCH 64          // number of chunks
#define DVB 64          // dv columns handled per workgroup

__device__ __forceinline__ __bf16 tobf(float f) {
  union { float f; unsigned u; } a; a.f = f;
  unsigned r = a.u + 0x7FFFu + ((a.u >> 16) & 1u);
  union { unsigned short u; __bf16 b; } c; c.u = (unsigned short)(r >> 16);
  return c.b;
}
__device__ __forceinline__ float frombf(__bf16 b) {
  union { unsigned short u; __bf16 b; } c; c.b = b;
  union { unsigned u; float f; } d; d.u = ((unsigned)c.u) << 16;
  return d.f;
}
__device__ __forceinline__ v8f zero8() {
  v8f z = {0.f,0.f,0.f,0.f,0.f,0.f,0.f,0.f};
  return z;
}
__device__ __forceinline__ v8f wmma_bf(v16bf a, v16bf b, v8f c) {
  return __builtin_amdgcn_wmma_f32_16x16x32_bf16(false, a, false, b, (short)0, c,
                                                 false, false);
}

// CDNA5 async copy global -> LDS (ASYNCcnt-tracked, no VGPR staging).
// lds_off: low 32 bits of the generic LDS address (= LDS byte offset).
__device__ __forceinline__ void async_b128(const void* lds_ptr, const void* gptr) {
  unsigned la = (unsigned)(uintptr_t)lds_ptr;
  unsigned long long ga = (unsigned long long)(uintptr_t)gptr;
  asm volatile("global_load_async_to_lds_b128 %0, %1, off"
               :: "v"(la), "v"(ga) : "memory");
}
__device__ __forceinline__ void async_b128_o16(const void* lds_ptr, const void* gptr) {
  unsigned la = (unsigned)(uintptr_t)lds_ptr;
  unsigned long long ga = (unsigned long long)(uintptr_t)gptr;
  asm volatile("global_load_async_to_lds_b128 %0, %1, off offset:16"
               :: "v"(la), "v"(ga) : "memory");
}

// Load a 16x32 bf16 fragment (A-style; also used for B via transposed storage).
// CDNA5 layout: lane L -> row L&15, K in {koff..koff+7, koff+16..+23}, koff=(L>=16)*8
__device__ __forceinline__ v16bf load_frag(const __bf16* base, int ld) {
  int lane = threadIdx.x & 31;
  int r = lane & 15;
  int koff = (lane >> 4) << 3;
  const __bf16* p = base + r * ld + koff;
  v8bf lo = *(const v8bf*)(p);
  v8bf hi = *(const v8bf*)(p + 16);
  v16bf f;
#pragma unroll
  for (int e = 0; e < 8; ++e) { f[e] = lo[e]; f[e + 8] = hi[e]; }
  return f;
}

__device__ __forceinline__ float waveRed(float v) {
#pragma unroll
  for (int o = 16; o; o >>= 1) v += __shfl_xor(v, o);
  return v;
}
__device__ __forceinline__ float blockReduce256(float v, float* red) {
  v = waveRed(v);
  __syncthreads();                 // guard red reuse across calls
  if ((threadIdx.x & 31) == 0) red[threadIdx.x >> 5] = v;
  __syncthreads();
  float tot = 0.f;
#pragma unroll
  for (int i = 0; i < 8; ++i) tot += red[i];
  return tot;
}

// ---------------------------------------------------------------------------
// elementwise converts
// ---------------------------------------------------------------------------
__global__ __launch_bounds__(256) void cvt_bf16_kernel(const float* __restrict__ in,
                                                       __bf16* __restrict__ out) {
  size_t i = (size_t)blockIdx.x * 256 + threadIdx.x;
  out[i] = tobf(in[i]);
}

// Wt[n*D + k] = W[k*D + n]  (bf16, transposed so GEMM B-fragments are row-major)
__global__ __launch_bounds__(256) void transW_kernel(const float* __restrict__ W,
                                                     __bf16* __restrict__ Wt) {
  size_t idx = (size_t)blockIdx.x * 256 + threadIdx.x;   // = k*D + n
  size_t k = idx >> 10, n = idx & 1023;
  Wt[n * GD + k] = tobf(W[idx]);
}

// ---------------------------------------------------------------------------
// bf16 WMMA GEMM: C[M,N] (f32) = A[M,K] (bf16, row-major) @ Bt[N,K]^T (bf16)
// 128x128 block tile, 256 threads = 8 waves in 4x2, each wave a 32x64 patch.
// Double-buffered LDS tiles filled by global_load_async_to_lds_b128 so the
// next k-slice streams in while WMMAs consume the current one.
// ---------------------------------------------------------------------------
#define GEMM_BM 128
#define GEMM_BN 128
#define GEMM_BK 32
#define GEMM_LDA (GEMM_BK + 8)

__global__ __launch_bounds__(256) void gemm_bf16_kernel(
    const __bf16* __restrict__ A, const __bf16* __restrict__ Bt,
    float* __restrict__ C, int M, int N, int K) {
  __shared__ __bf16 As[2][GEMM_BM][GEMM_LDA];
  __shared__ __bf16 Bs[2][GEMM_BN][GEMM_LDA];
  const int tid = threadIdx.x;
  const int wave = tid >> 5, lane = tid & 31;
  const int col = lane & 15, mrow0 = (lane >> 4) << 3;
  const int wm = wave & 3, wn = wave >> 2;
  const int m0 = blockIdx.x * GEMM_BM;
  const int n0 = blockIdx.y * GEMM_BN;

  v8f acc[2][4];
#pragma unroll
  for (int mi = 0; mi < 2; ++mi)
#pragma unroll
    for (int nt = 0; nt < 4; ++nt) acc[mi][nt] = zero8();

  // each thread owns 32 bytes (2 x b128) of one row in each tile
  const int lrow = tid >> 1, le0 = (tid & 1) << 4;
  const __bf16* ag = A  + (size_t)(m0 + lrow) * K + le0;
  const __bf16* bg = Bt + (size_t)(n0 + lrow) * K + le0;

  auto issue = [&](int bufi, int k0) {
    async_b128(&As[bufi][lrow][le0], ag + k0);
    async_b128_o16(&As[bufi][lrow][le0], ag + k0);
    async_b128(&Bs[bufi][lrow][le0], bg + k0);
    async_b128_o16(&Bs[bufi][lrow][le0], bg + k0);
  };

  issue(0, 0);
  int buf = 0;
  for (int k0 = 0; k0 < K; k0 += GEMM_BK) {
    if (k0 + GEMM_BK < K) {
      issue(buf ^ 1, k0 + GEMM_BK);
      asm volatile("s_wait_asynccnt 4" ::: "memory");  // current tile landed
    } else {
      asm volatile("s_wait_asynccnt 0" ::: "memory");
    }
    __syncthreads();
    v16bf af0 = load_frag(&As[buf][wm * 32][0], GEMM_LDA);
    v16bf af1 = load_frag(&As[buf][wm * 32 + 16][0], GEMM_LDA);
#pragma unroll
    for (int nt = 0; nt < 4; ++nt) {
      v16bf bf = load_frag(&Bs[buf][wn * 64 + nt * 16][0], GEMM_LDA);
      acc[0][nt] = wmma_bf(af0, bf, acc[0][nt]);
      acc[1][nt] = wmma_bf(af1, bf, acc[1][nt]);
    }
    __syncthreads();
    buf ^= 1;
  }
#pragma unroll
  for (int mi = 0; mi < 2; ++mi)
#pragma unroll
    for (int nt = 0; nt < 4; ++nt)
#pragma unroll
      for (int i = 0; i < 8; ++i) {
        int m = m0 + wm * 32 + mi * 16 + mrow0 + i;
        int n = n0 + wn * 64 + nt * 16 + col;
        C[(size_t)m * N + n] = acc[mi][nt][i];
      }
}

// ---------------------------------------------------------------------------
// per-(b,t) q/k L2 norm -> bf16 (q also * DK^-0.5)
// ---------------------------------------------------------------------------
__global__ __launch_bounds__(256) void normqk_kernel(
    const float* __restrict__ q, const float* __restrict__ k,
    __bf16* __restrict__ qo, __bf16* __restrict__ ko) {
  __shared__ float red[8];
  size_t row = blockIdx.x;
  int tid = threadIdx.x;
  for (int h = 0; h < GH; ++h) {
    size_t base = row * GD + (size_t)h * GDK + tid;
    float qv = q[base], kv = k[base];
    float qs = blockReduce256(qv * qv, red);
    float ks = blockReduce256(kv * kv, red);
    qo[base] = tobf(qv * rsqrtf(qs + 1e-6f) * 0.0625f);  // 256^-0.5
    ko[base] = tobf(kv * rsqrtf(ks + 1e-6f));
  }
}

// ---------------------------------------------------------------------------
// beta = sigmoid(x@Wb), dec = exp(-exp(A_log)*softplus(x@Wa + dt_bias))
// ---------------------------------------------------------------------------
__global__ __launch_bounds__(256) void betadec_kernel(
    const float* __restrict__ x, const float* __restrict__ Wb,
    const float* __restrict__ Wa, const float* __restrict__ A_log,
    const float* __restrict__ dt_bias, float* __restrict__ beta,
    float* __restrict__ dec) {
  __shared__ float red[8][8];
  size_t row = blockIdx.x;
  int tid = threadIdx.x;
  float pb[4] = {0, 0, 0, 0}, pa[4] = {0, 0, 0, 0};
  const float* xr = x + row * GD;
  for (int kk = tid; kk < GD; kk += 256) {
    float xv = xr[kk];
#pragma unroll
    for (int hh = 0; hh < 4; ++hh) {
      pb[hh] = fmaf(xv, Wb[kk * 4 + hh], pb[hh]);
      pa[hh] = fmaf(xv, Wa[kk * 4 + hh], pa[hh]);
    }
  }
#pragma unroll
  for (int hh = 0; hh < 4; ++hh) {
    float bsum = waveRed(pb[hh]);
    float asum = waveRed(pa[hh]);
    if ((tid & 31) == 0) { red[tid >> 5][hh] = bsum; red[tid >> 5][4 + hh] = asum; }
  }
  __syncthreads();
  if (tid < 8) {
    float s = 0.f;
#pragma unroll
    for (int w = 0; w < 8; ++w) s += red[w][tid];
    if (tid < 4) {
      beta[row * 4 + tid] = 1.f / (1.f + expf(-s));
    } else {
      int hh = tid - 4;
      float z = s + dt_bias[hh];
      float sp = (z > 20.f) ? z : log1pf(expf(z));
      dec[row * 4 + hh] = expf(-expf(A_log[hh]) * sp);
    }
  }
}

// ---------------------------------------------------------------------------
// chunked gated delta-rule recurrence.
// grid = 64 blocks: (b,h,vb) with vb a 64-column block of DV.
// LDS-resident f32 state slab S[256][64] + bf16 transposed mirror (~270KB,
// feasible only because gfx1250 has 320KB LDS per WGP).
// ---------------------------------------------------------------------------
#define S_SF 66
#define S_KQ 264
#define S_KT 72
#define S_AF 66
#define S_MB 72
#define S_BT 67
#define S_UF 67
#define S_UB 72

#define OFF_SF  ((size_t)0)
#define OFF_SBT (OFF_SF  + (size_t)256 * S_SF * 4)
#define OFF_KC  (OFF_SBT + (size_t)64 * S_KQ * 2)
#define OFF_QC  (OFF_KC  + (size_t)64 * S_KQ * 2)
#define OFF_KT  (OFF_QC  + (size_t)64 * S_KQ * 2)
#define OFF_AF  (OFF_KT  + (size_t)256 * S_KT * 2)
#define OFF_MB  (OFF_AF  + (size_t)64 * S_AF * 4)
#define OFF_BT  (OFF_MB  + (size_t)64 * S_MB * 2)
#define OFF_UF  (OFF_BT  + (size_t)64 * S_BT * 4)
#define OFF_UB  (OFF_UF  + (size_t)64 * S_UF * 4)
#define OFF_LG  (OFF_UB  + (size_t)64 * S_UB * 2)
#define OFF_BE  (OFF_LG  + (size_t)64 * 4)
#define SMEM_CHUNK (OFF_BE + (size_t)64 * 4)         // ~270 KB < 320 KB/WGP

__global__ __launch_bounds__(256) void gdn_chunk_kernel(
    const __bf16* __restrict__ qbf, const __bf16* __restrict__ kbf,
    const float* __restrict__ vg, const float* __restrict__ betag,
    const float* __restrict__ decg, float* __restrict__ og) {
  extern __shared__ char smem_raw[];
  float*  Sf  = (float*)(smem_raw + OFF_SF);    // S[dk][dv]     f32 master
  __bf16* SbT = (__bf16*)(smem_raw + OFF_SBT);  // S^T[dv][dk]   bf16 mirror
  __bf16* Kc  = (__bf16*)(smem_raw + OFF_KC);   // K[t][dk]
  __bf16* Qc  = (__bf16*)(smem_raw + OFF_QC);   // Q[t][dk]
  __bf16* KT  = (__bf16*)(smem_raw + OFF_KT);   // (gL/g_s)*K^T[dk][s]
  float*  Af  = (float*)(smem_raw + OFF_AF);    // A[t][s] strict lower
  __bf16* Mb  = (__bf16*)(smem_raw + OFF_MB);   // M[t][s] incl lower
  float*  bT  = (float*)(smem_raw + OFF_BT);    // b^T[dv][t]
  float*  Uf  = (float*)(smem_raw + OFF_UF);    // U^T[dv][s]    f32
  __bf16* Ub  = (__bf16*)(smem_raw + OFF_UB);   // U^T[dv][s]    bf16
  float*  lg  = (float*)(smem_raw + OFF_LG);    // cumulative log-decay
  float*  be  = (float*)(smem_raw + OFF_BE);    // beta per step

  const int tid = threadIdx.x;
  const int wave = tid >> 5, lane = tid & 31;
  const int col = lane & 15, mrow0 = (lane >> 4) << 3;
  const int b = blockIdx.x >> 4, h = (blockIdx.x >> 2) & 3, vb = blockIdx.x & 3;
  const int mt = wave >> 1, ntb = (wave & 1) << 1;  // output tiles for 64x64 GEMMs

  // zero state
  for (int idx = tid; idx < 256 * DVB; idx += 256) {
    int m = idx >> 6, n = idx & 63;
    Sf[m * S_SF + n] = 0.f;
    SbT[n * S_KQ + m] = tobf(0.f);
  }
  __syncthreads();

  for (int c = 0; c < NCH; ++c) {
    const int t0 = c * CHL;
    // ---- stage 1: async-load K,Q chunk tiles straight into LDS
    for (int idx = tid; idx < CHL * 32; idx += 256) {
      int r = idx >> 5, ko = (idx & 31) << 3;
      size_t g = (((size_t)(b * GT + t0 + r) * GH + h) << 8) + ko;
      async_b128(&Kc[r * S_KQ + ko], kbf + g);
      async_b128(&Qc[r * S_KQ + ko], qbf + g);
    }
    if (tid < CHL) {
      size_t g = (size_t)(b * GT + t0 + tid) * GH + h;
      be[tid] = betag[g];
      lg[tid] = logf(decg[g]);
    }
    asm volatile("s_wait_asynccnt 0" ::: "memory");
    __syncthreads();
    // ---- stage 2: inclusive prefix sum of log-decay
    if (tid == 0) {
      float run = 0.f;
      for (int t = 0; t < CHL; ++t) { run += lg[t]; lg[t] = run; }
    }
    __syncthreads();
    const float lgL = lg[CHL - 1];
    // ---- stage 3: transposed, tail-decayed K
    for (int idx = tid; idx < 256 * CHL; idx += 256) {
      int kk = idx >> 6, s = idx & 63;
      KT[kk * S_KT + s] = tobf(frombf(Kc[s * S_KQ + kk]) * expf(lgL - lg[s]));
    }
    __syncthreads();
    // ---- stage 4: three WMMA GEMMs -> Af, Mb, bT
    {
      // G1: K K^T
      v8f a0 = zero8(), a1 = zero8();
#pragma unroll
      for (int kk = 0; kk < GDK; kk += 32) {
        v16bf fA = load_frag(Kc + mt * 16 * S_KQ + kk, S_KQ);
        a0 = wmma_bf(fA, load_frag(Kc + ntb * 16 * S_KQ + kk, S_KQ), a0);
        a1 = wmma_bf(fA, load_frag(Kc + (ntb + 1) * 16 * S_KQ + kk, S_KQ), a1);
      }
#pragma unroll
      for (int j = 0; j < 2; ++j) {
        v8f a = j ? a1 : a0;
        int nt = ntb + j;
#pragma unroll
        for (int i = 0; i < 8; ++i) {
          int m = mt * 16 + mrow0 + i, n = nt * 16 + col;
          Af[m * S_AF + n] = (n < m) ? be[m] * expf(lg[m] - lg[n]) * a[i] : 0.f;
        }
      }
      // G2: Q K^T
      a0 = zero8(); a1 = zero8();
#pragma unroll
      for (int kk = 0; kk < GDK; kk += 32) {
        v16bf fA = load_frag(Qc + mt * 16 * S_KQ + kk, S_KQ);
        a0 = wmma_bf(fA, load_frag(Kc + ntb * 16 * S_KQ + kk, S_KQ), a0);
        a1 = wmma_bf(fA, load_frag(Kc + (ntb + 1) * 16 * S_KQ + kk, S_KQ), a1);
      }
#pragma unroll
      for (int j = 0; j < 2; ++j) {
        v8f a = j ? a1 : a0;
        int nt = ntb + j;
#pragma unroll
        for (int i = 0; i < 8; ++i) {
          int m = mt * 16 + mrow0 + i, n = nt * 16 + col;
          Mb[m * S_MB + n] = tobf((n <= m) ? expf(lg[m] - lg[n]) * a[i] : 0.f);
        }
      }
      // G3: K @ S0   ->  b^T = beta*(v - gamma*pred)
      a0 = zero8(); a1 = zero8();
#pragma unroll
      for (int kk = 0; kk < GDK; kk += 32) {
        v16bf fA = load_frag(Kc + mt * 16 * S_KQ + kk, S_KQ);
        a0 = wmma_bf(fA, load_frag(SbT + ntb * 16 * S_KQ + kk, S_KQ), a0);
        a1 = wmma_bf(fA, load_frag(SbT + (ntb + 1) * 16 * S_KQ + kk, S_KQ), a1);
      }
#pragma unroll
      for (int j = 0; j < 2; ++j) {
        v8f a = j ? a1 : a0;
        int nt = ntb + j;
#pragma unroll
        for (int i = 0; i < 8; ++i) {
          int m = mt * 16 + mrow0 + i, n = nt * 16 + col;
          size_t gvi = ((size_t)(b * GT + t0 + m) * GH + h) * GDV + vb * DVB + n;
          bT[n * S_BT + m] = be[m] * (vg[gvi] - expf(lg[m]) * a[i]);
        }
      }
    }
    __syncthreads();
    // ---- stage 5: forward substitution  (I+A) U = b  (per dv column)
    if (tid < CHL) {
      const int cdv = tid;
      for (int t = 0; t < CHL; ++t) {
        float acc = bT[cdv * S_BT + t];
        for (int s = 0; s < t; ++s) acc -= Af[t * S_AF + s] * Uf[cdv * S_UF + s];
        Uf[cdv * S_UF + t] = acc;
      }
    }
    __syncthreads();
    for (int idx = tid; idx < CHL * CHL; idx += 256) {
      int r = idx >> 6, s = idx & 63;
      Ub[r * S_UB + s] = tobf(Uf[r * S_UF + s]);
    }
    __syncthreads();
    // ---- stage 6: O = diag(gamma) Q S0 + M U  -> global o
    {
      v8f a0 = zero8(), a1 = zero8();
#pragma unroll
      for (int kk = 0; kk < GDK; kk += 32) {
        v16bf fA = load_frag(Qc + mt * 16 * S_KQ + kk, S_KQ);
        a0 = wmma_bf(fA, load_frag(SbT + ntb * 16 * S_KQ + kk, S_KQ), a0);
        a1 = wmma_bf(fA, load_frag(SbT + (ntb + 1) * 16 * S_KQ + kk, S_KQ), a1);
      }
#pragma unroll
      for (int i = 0; i < 8; ++i) {
        float gm = expf(lg[mt * 16 + mrow0 + i]);
        a0[i] *= gm; a1[i] *= gm;
      }
#pragma unroll
      for (int kk = 0; kk < CHL; kk += 32) {
        v16bf fA = load_frag(Mb + mt * 16 * S_MB + kk, S_MB);
        a0 = wmma_bf(fA, load_frag(Ub + ntb * 16 * S_UB + kk, S_UB), a0);
        a1 = wmma_bf(fA, load_frag(Ub + (ntb + 1) * 16 * S_UB + kk, S_UB), a1);
      }
#pragma unroll
      for (int j = 0; j < 2; ++j) {
        v8f a = j ? a1 : a0;
        int nt = ntb + j;
#pragma unroll
        for (int i = 0; i < 8; ++i) {
          int m = mt * 16 + mrow0 + i, n = nt * 16 + col;
          og[((size_t)(b * GT + t0 + m) * GH + h) * GDV + vb * DVB + n] = a[i];
        }
      }
    }
    // ---- stage 7: S = gammaL * S + K~^T U  (64 tiles of 16x16, 8 per wave)
    {
      const float gL = expf(lgL);
#pragma unroll
      for (int j = 0; j < 8; ++j) {
        int tile = wave * 8 + j;
        int smt = tile >> 2, snt = tile & 3;
        v8f sa = zero8();
#pragma unroll
        for (int kk = 0; kk < CHL; kk += 32) {
          v16bf fA = load_frag(KT + smt * 16 * S_KT + kk, S_KT);
          sa = wmma_bf(fA, load_frag(Ub + snt * 16 * S_UB + kk, S_UB), sa);
        }
#pragma unroll
        for (int i = 0; i < 8; ++i) {
          int m = smt * 16 + mrow0 + i, n = snt * 16 + col;
          Sf[m * S_SF + n] = gL * Sf[m * S_SF + n] + sa[i];
        }
      }
    }
    __syncthreads();
    // ---- stage 8: refresh bf16 transposed mirror of S
    for (int idx = tid; idx < 256 * DVB; idx += 256) {
      int m = idx >> 6, n = idx & 63;
      SbT[n * S_KQ + m] = tobf(Sf[m * S_SF + n]);
    }
    __syncthreads();
  }
}

// ---------------------------------------------------------------------------
// gated RMSNorm epilogue: o * rsqrt(mean(o^2)+1e-5) * norm_w * silu(gate) -> bf16
// ---------------------------------------------------------------------------
__global__ __launch_bounds__(256) void normgate_kernel(
    const float* __restrict__ o, const float* __restrict__ g,
    const float* __restrict__ nw, __bf16* __restrict__ out) {
  __shared__ float red[8];
  size_t row = blockIdx.x;
  int tid = threadIdx.x;
  for (int hh = 0; hh < GH; ++hh) {
    size_t base = row * GD + (size_t)hh * GDV + tid;
    float ov = o[base];
    float ms = blockReduce256(ov * ov, red) * (1.f / 256.f);
    float gv = g[base];
    float sil = gv / (1.f + expf(-gv));
    out[base] = tobf(ov * rsqrtf(ms + 1e-5f) * nw[tid] * sil);
  }
}

// ---------------------------------------------------------------------------
// launch
// ---------------------------------------------------------------------------
extern "C" void kernel_launch(void* const* d_in, const int* in_sizes, int n_in,
                              void* d_out, int out_size, void* d_ws, size_t ws_size,
                              hipStream_t stream) {
  (void)in_sizes; (void)n_in; (void)out_size; (void)ws_size;
  const float* x       = (const float*)d_in[0];
  const float* Wq      = (const float*)d_in[1];
  const float* Wk      = (const float*)d_in[2];
  const float* Wv      = (const float*)d_in[3];
  const float* Wb      = (const float*)d_in[4];
  const float* Wa      = (const float*)d_in[5];
  const float* A_log   = (const float*)d_in[6];
  const float* dt_bias = (const float*)d_in[7];
  const float* Wg      = (const float*)d_in[8];
  const float* norm_w  = (const float*)d_in[9];
  const float* Wo      = (const float*)d_in[10];
  float* out = (float*)d_out;

  const size_t NX = (size_t)GB * GT * GD;   // 16.8M elements
  const size_t NW = (size_t)GD * GD;

  char* ws = (char*)d_ws;
  size_t off = 0;
  auto alloc = [&](size_t bytes) -> void* {
    void* p = ws + off;
    off += (bytes + 255) & ~(size_t)255;
    return p;
  };
  __bf16* xbf  = (__bf16*)alloc(NX * 2);
  __bf16* Wqt  = (__bf16*)alloc(NW * 2);
  __bf16* Wkt  = (__bf16*)alloc(NW * 2);
  __bf16* Wvt  = (__bf16*)alloc(NW * 2);
  __bf16* Wgt  = (__bf16*)alloc(NW * 2);
  __bf16* Wot  = (__bf16*)alloc(NW * 2);
  float*  qraw = (float*)alloc(NX * 4);
  float*  kraw = (float*)alloc(NX * 4);
  float*  vbuf = (float*)alloc(NX * 4);
  float*  gbuf = (float*)alloc(NX * 4);
  __bf16* qbf  = (__bf16*)alloc(NX * 2);
  __bf16* kbf  = (__bf16*)alloc(NX * 2);
  float*  beta = (float*)alloc((size_t)GB * GT * GH * 4);
  float*  dec  = (float*)alloc((size_t)GB * GT * GH * 4);
  // lifetime reuse: qraw/kraw are dead after normqk -> reuse for o / o_norm_bf16
  float*  obuf = qraw;
  __bf16* obf  = (__bf16*)kraw;

  const int ROWS = GB * GT;                 // 16384

  // 1) input / weight conversion
  cvt_bf16_kernel<<<dim3((unsigned)(NX / 256)), dim3(256), 0, stream>>>(x, xbf);
  transW_kernel<<<dim3((unsigned)(NW / 256)), dim3(256), 0, stream>>>(Wq, Wqt);
  transW_kernel<<<dim3((unsigned)(NW / 256)), dim3(256), 0, stream>>>(Wk, Wkt);
  transW_kernel<<<dim3((unsigned)(NW / 256)), dim3(256), 0, stream>>>(Wv, Wvt);
  transW_kernel<<<dim3((unsigned)(NW / 256)), dim3(256), 0, stream>>>(Wg, Wgt);
  transW_kernel<<<dim3((unsigned)(NW / 256)), dim3(256), 0, stream>>>(Wo, Wot);

  // 2) projections (WMMA bf16 GEMMs, async double-buffered)
  dim3 ggrid(ROWS / GEMM_BM, GD / GEMM_BN);
  gemm_bf16_kernel<<<ggrid, 256, 0, stream>>>(xbf, Wqt, qraw, ROWS, GD, GD);
  gemm_bf16_kernel<<<ggrid, 256, 0, stream>>>(xbf, Wkt, kraw, ROWS, GD, GD);
  gemm_bf16_kernel<<<ggrid, 256, 0, stream>>>(xbf, Wvt, vbuf, ROWS, GD, GD);
  gemm_bf16_kernel<<<ggrid, 256, 0, stream>>>(xbf, Wgt, gbuf, ROWS, GD, GD);

  // 3) per-token scalars + q/k normalization
  normqk_kernel<<<dim3(ROWS), dim3(256), 0, stream>>>(qraw, kraw, qbf, kbf);
  betadec_kernel<<<dim3(ROWS), dim3(256), 0, stream>>>(x, Wb, Wa, A_log, dt_bias,
                                                       beta, dec);

  // 4) chunked gated delta-rule recurrence (64 WGs, ~270KB LDS each)
  gdn_chunk_kernel<<<dim3(GB * GH * 4), dim3(256), (unsigned)SMEM_CHUNK, stream>>>(
      qbf, kbf, vbuf, beta, dec, obuf);

  // 5) gated RMSNorm epilogue + final projection
  normgate_kernel<<<dim3(ROWS), dim3(256), 0, stream>>>(obuf, gbuf, norm_w, obf);
  gemm_bf16_kernel<<<ggrid, 256, 0, stream>>>(obf, Wot, out, ROWS, GD, GD);
}
